// RnnEncoder_27676769255964
// MI455X (gfx1250) — matrix operand order repaired and checked by reading, not compiled, not run on hardware
//
#include <hip/hip_runtime.h>
#include <hip/hip_bf16.h>

// Shapes from the reference
#define BB 128
#define TT 2048
#define FF 128
#define HH 256

typedef __attribute__((ext_vector_type(16))) __bf16 v16bf;
typedef __attribute__((ext_vector_type(8)))  __bf16 v8bf;
typedef __attribute__((ext_vector_type(4)))  __bf16 v4bf;
typedef __attribute__((ext_vector_type(8)))  float  v8f;

// Scheduling fence: nothing may be reordered across it by the machine scheduler
#if defined(__has_builtin)
#if __has_builtin(__builtin_amdgcn_sched_barrier)
#define SCHED_FENCE() __builtin_amdgcn_sched_barrier(0)
#endif
#endif
#ifndef SCHED_FENCE
#define SCHED_FENCE()
#endif

// LDS layout (bytes), padded rows to rotate banks (strides are multiples of 16B)
#define WHH_STRIDE 264   // 256 + 8 halves
#define WIH_STRIDE 136   // 128 + 8 halves
#define OFF_WHH   0                         // 256*264*2 = 135168
#define OFF_WIH   135168                    // 256*136*2 = 69632
#define OFF_H     204800                    // 16*264*2  = 8448
#define OFF_X     213248                    // 2*16*136*2= 8704
#define OFF_BIAS  221952                    // 256*4     = 1024
#define OFF_WFC   222976                    // 256*4     = 1024
#define OFF_LIN   224000                    // 16*4
#define OFF_LEN   224064                    // 16*4
#define SMEM_BYTES 224128

__device__ __forceinline__ v16bf cat16(v8bf lo, v8bf hi) {
  return __builtin_shufflevector(lo, hi, 0,1,2,3,4,5,6,7,8,9,10,11,12,13,14,15);
}

__device__ __forceinline__ v8f wmma_bf16(v16bf a, v16bf b, v8f c) {
  return __builtin_amdgcn_wmma_f32_16x16x32_bf16(false, a, false, b, (short)0, c,
                                                 false, false);
}

__device__ __forceinline__ float tanh_fast(float v) {
  v = __builtin_amdgcn_fmed3f(v, -10.0f, 10.0f);            // single v_med3_f32
  float e = __builtin_amdgcn_exp2f(v * 2.885390081777927f); // exp(2v)
  return (e - 1.0f) * __builtin_amdgcn_rcpf(e + 1.0f);
}

__global__ __launch_bounds__(512, 1)
void rnn_encoder_kernel(const float* __restrict__ x,
                        const int*   __restrict__ lengths,
                        const float* __restrict__ W_ih,
                        const float* __restrict__ W_hh,
                        const float* __restrict__ b_ih,
                        const float* __restrict__ b_hh,
                        const float* __restrict__ W_fc,
                        const float* __restrict__ b_fc,
                        float* __restrict__ out) {
  extern __shared__ char smem[];
  __bf16* sWhh = (__bf16*)(smem + OFF_WHH);   // [256][264] W_hh[n][k] (bf16)
  __bf16* sWih = (__bf16*)(smem + OFF_WIH);   // [256][136] W_ih[n][k] (bf16)
  __bf16* sH   = (__bf16*)(smem + OFF_H);     // [16][264]  h_{t-1} (bf16)
  __bf16* sX   = (__bf16*)(smem + OFF_X);     // [2][16][136] x tile (bf16)
  float*  sBias= (float*)(smem + OFF_BIAS);   // b_ih + b_hh
  float*  sWfc = (float*)(smem + OFF_WFC);
  float*  sLin = (float*)(smem + OFF_LIN);    // per-row fc accumulator
  int*    sLen = (int*)(smem + OFF_LEN);

  const int tid  = threadIdx.x;
  const int lane = tid & 31;
  const int wave = tid >> 5;                 // 16 waves, one N-tile each
  const int b0   = blockIdx.x * 16;          // 8 blocks x 16 batch rows

  // ---- stage weights into LDS (fp32 -> bf16), once ----
  for (int i = tid; i < HH * HH; i += 512)
    sWhh[(i >> 8) * WHH_STRIDE + (i & 255)] = (__bf16)W_hh[i];
  for (int i = tid; i < HH * FF; i += 512)
    sWih[(i >> 7) * WIH_STRIDE + (i & 127)] = (__bf16)W_ih[i];
  if (tid < HH) { sBias[tid] = b_ih[tid] + b_hh[tid]; sWfc[tid] = W_fc[tid]; }
  for (int i = tid; i < 16 * WHH_STRIDE; i += 512) sH[i] = (__bf16)0.0f;
  if (tid < 16) { sLen[tid] = lengths[b0 + tid]; sLin[tid] = 0.0f; }
  const float bfc = b_fc[0];

  const int halfSel = lane >> 4;            // 0: lanes 0-15, 1: lanes 16-31
  const int nl = lane & 15;
  const int n = wave * 16 + nl;             // this wave's N column

  // ---- x streaming: each thread owns 4 floats of the 16x128 tile ----
  const int xm  = tid >> 5;
  const int xf0 = (tid & 31) << 2;
  const float* xsrc = x + ((size_t)(b0 + xm) * TT) * FF + xf0;  // + t*FF per step
  __bf16* xdst0 = &sX[(0 * 16 + xm) * WIH_STRIDE + xf0];
  __bf16* xdst1 = &sX[(1 * 16 + xm) * WIH_STRIDE + xf0];

  auto stage_x = [&](float4 u, int buf) {   // cvt + ds_store (no global wait here)
    v4bf o;
    o[0]=(__bf16)u.x; o[1]=(__bf16)u.y; o[2]=(__bf16)u.z; o[3]=(__bf16)u.w;
    *(v4bf*)(buf ? xdst1 : xdst0) = o;
  };

  // pipeline prologue: stage t=0, issue load of t=1
  stage_x(*(const float4*)(xsrc + 0 * FF), 0);
  float4 xreg = *(const float4*)(xsrc + 1 * FF);
  __builtin_prefetch(xsrc + 2 * FF, 0, 0);  // global_prefetch_b8
  __syncthreads();

  // ---- hoist this wave's B operands (weights) into VGPRs, once ----
  v16bf wIh[4];
  #pragma unroll
  for (int kc = 0; kc < 4; ++kc) {
    const __bf16* bp = &sWih[n * WIH_STRIDE + kc * 32 + halfSel * 8];
    wIh[kc] = cat16(*(const v8bf*)bp, *(const v8bf*)(bp + 16));
  }
  v16bf wHh[8];
  #pragma unroll
  for (int kc = 0; kc < 8; ++kc) {
    const __bf16* bp = &sWhh[n * WHH_STRIDE + kc * 32 + halfSel * 8];
    wHh[kc] = cat16(*(const v8bf*)bp, *(const v8bf*)(bp + 16));
  }

  int myLen[8];
  #pragma unroll
  for (int r = 0; r < 8; ++r) myLen[r] = sLen[r + halfSel * 8];
  const float wfc_n = sWfc[n];
  const float bias_n = sBias[n];

  int cnt = 0;
  float* outH = out + BB;                    // hiddens follow counts

  for (int t = 0; t < TT; ++t) {
    const int buf = t & 1;

    // stage x_{t+1} (loaded a full step ago -> loadcnt wait is free),
    // then issue the load for x_{t+2}; pin this block at the top of the step
    if (t + 1 < TT) {
      stage_x(xreg, buf ^ 1);
      if (t + 2 < TT) {
        xreg = *(const float4*)(xsrc + (size_t)(t + 2) * FF);
        __builtin_prefetch(xsrc + (size_t)(t + 3) * FF, 0, 0);
      }
    }
    SCHED_FENCE();   // keep the global load issued early, decoupled from DS waits

    // ---- preload ALL A operands into distinct registers; fence keeps the
    //      24 ds_load_b128 in flight ahead of the WMMA chain ----
    v16bf aX[4];
    #pragma unroll
    for (int kc = 0; kc < 4; ++kc) {
      const __bf16* ap = &sX[(buf * 16 + nl) * WIH_STRIDE + kc * 32 + halfSel * 8];
      aX[kc] = cat16(*(const v8bf*)ap, *(const v8bf*)(ap + 16));
    }
    v16bf aH[8];
    #pragma unroll
    for (int kc = 0; kc < 8; ++kc) {
      const __bf16* ap = &sH[nl * WHH_STRIDE + kc * 32 + halfSel * 8];
      aH[kc] = cat16(*(const v8bf*)ap, *(const v8bf*)(ap + 16));
    }
    SCHED_FENCE();   // loads may not sink past here; WMMAs may not hoist above

    // two independent accumulation chains (6 WMMAs each) for matrix-pipe ILP
    v8f acc0, acc1 = {};
    #pragma unroll
    for (int i = 0; i < 8; ++i) acc0[i] = bias_n;
    #pragma unroll
    for (int kc = 0; kc < 4; ++kc) {
      if (kc & 1) acc1 = wmma_bf16(aX[kc], wIh[kc], acc1);
      else        acc0 = wmma_bf16(aX[kc], wIh[kc], acc0);
    }
    #pragma unroll
    for (int kc = 0; kc < 8; ++kc) {
      if (kc & 1) acc1 = wmma_bf16(aH[kc], wHh[kc], acc1);
      else        acc0 = wmma_bf16(aH[kc], wHh[kc], acc0);
    }
    v8f acc;
    #pragma unroll
    for (int i = 0; i < 8; ++i) acc[i] = acc0[i] + acc1[i];

    __syncthreads();   // all reads of sH done before overwrite

    // tanh, commit h_t, fc partials
    float hvv[8];
    float pr[8];
    bool anyFinal = false;
    #pragma unroll
    for (int r = 0; r < 8; ++r) {
      const int m = r + halfSel * 8;        // C/D layout: VGPR r, half -> row
      const float hv = tanh_fast(acc[r]);
      hvv[r] = hv;
      sH[m * WHH_STRIDE + n] = (__bf16)hv;
      pr[r] = hv * wfc_n;
      anyFinal |= (t == myLen[r] - 1);
    }
    // cold path: final-hidden store fires once per row over the whole T loop
    if (__builtin_expect(anyFinal, 0)) {
      #pragma unroll
      for (int r = 0; r < 8; ++r)
        if (t == myLen[r] - 1)
          outH[(size_t)(b0 + r + halfSel * 8) * HH + n] = hvv[r];
    }
    // reduce fc partials over the 16 lanes of each half, then LDS-accumulate
    #pragma unroll
    for (int r = 0; r < 8; ++r) {
      float s = pr[r];
      s += __shfl_xor(s, 1, 32);
      s += __shfl_xor(s, 2, 32);
      s += __shfl_xor(s, 4, 32);
      s += __shfl_xor(s, 8, 32);
      if (nl == 0) atomicAdd(&sLin[r + halfSel * 8], s);  // ds_add_f32
    }
    __syncthreads();   // sLin complete; sH visible for next step

    if (tid < 16) {    // wave 0: threshold, count, reset
      const float lin = sLin[tid] + bfc;
      if (lin > 0.0f && t < sLen[tid]) ++cnt;
      sLin[tid] = 0.0f;
    }
  }

  __syncthreads();
  if (tid < 16) out[b0 + tid] = (float)cnt;   // counts (B,1) as output dtype
}

extern "C" void kernel_launch(void* const* d_in, const int* in_sizes, int n_in,
                              void* d_out, int out_size, void* d_ws, size_t ws_size,
                              hipStream_t stream) {
  const float* x      = (const float*)d_in[0];
  const int*   len    = (const int*)  d_in[1];
  const float* W_ih   = (const float*)d_in[2];
  const float* W_hh   = (const float*)d_in[3];
  const float* b_ih   = (const float*)d_in[4];
  const float* b_hh   = (const float*)d_in[5];
  const float* W_fc   = (const float*)d_in[6];
  const float* b_fc   = (const float*)d_in[7];
  float* out = (float*)d_out;

  rnn_encoder_kernel<<<dim3(BB / 16), dim3(512), SMEM_BYTES, stream>>>(
      x, len, W_ih, W_hh, b_ih, b_hh, W_fc, b_fc, out);
}
